// PointLocalScaleAttention_44770739093793
// MI455X (gfx1250) — compile-verified, matrix operand order checked by reading, not run on hardware
//
#include <hip/hip_runtime.h>
#include <hip/hip_bf16.h>

// ---------------- problem constants ----------------
#define Bx   4
#define Cc   256
#define Nn   8192
#define Kk   16
#define Hh   8
#define HCc  32
#define CSs  128
#define HSs  4
#define SCALE_F 0.17677669529663687f   // 32^-0.5
#define EPS_F   1e-5f

typedef __bf16 bf16_t;
typedef __attribute__((ext_vector_type(16))) __bf16 v16bf;
typedef __attribute__((ext_vector_type(8)))  float  v8f;

// ---------------- helpers ----------------
__device__ __forceinline__ bf16_t f2bf(float x) {
  unsigned u = __builtin_bit_cast(unsigned, x);
  unsigned r = u + 0x7fffu + ((u >> 16) & 1u);   // round-to-nearest-even
  unsigned short h = (unsigned short)(r >> 16);
  return __builtin_bit_cast(bf16_t, h);
}

__device__ __forceinline__ v8f zero8() {
  v8f a;
#pragma unroll
  for (int r = 0; r < 8; ++r) a[r] = 0.0f;
  return a;
}

// CDNA5 async copy: 16 bytes global -> LDS per lane, tracked on ASYNCcnt.
__device__ __forceinline__ void async_copy16(unsigned lds_off, const void* base,
                                             unsigned byte_off) {
  asm volatile("global_load_async_to_lds_b128 %0, %1, %2 offset:0"
               :: "v"(lds_off), "v"(byte_off), "s"(base)
               : "memory");
}
__device__ __forceinline__ void wait_async0() {
  asm volatile("s_wait_asynccnt 0x0" ::: "memory");
}

// A fragment: 16x32 bf16 tile, row-major source (row = M, col = contraction K).
// Layout per ISA 7.12.2: lanes 0-15 M=0..15 K-groups {0-7,16-23}, lanes 16-31 {8-15,24-31}.
__device__ __forceinline__ v16bf frag_a(const bf16_t* base, int ld, int kbase) {
  int lane = threadIdx.x & 31;
  int m = lane & 15;
  int koff = (lane & 16) ? 8 : 0;
  const bf16_t* p = base + (size_t)m * ld + kbase + koff;
  v16bf f;
#pragma unroll
  for (int r = 0; r < 4; ++r) {
    f[2*r]     = p[2*r];
    f[2*r+1]   = p[2*r+1];
    f[8+2*r]   = p[16 + 2*r];
    f[8+2*r+1] = p[16 + 2*r+1];
  }
  return f;
}

// B fragment for B[k,n] = W[rowbase + n*rowstep][kbase + k], W row-major [out][in].
__device__ __forceinline__ v16bf frag_b(const bf16_t* w, int ld, int rowbase,
                                        int rowstep, int kbase) {
  int lane = threadIdx.x & 31;
  int n = lane & 15;
  int koff = (lane & 16) ? 8 : 0;
  const bf16_t* p = w + (size_t)(rowbase + n * rowstep) * ld + kbase + koff;
  v16bf f;
#pragma unroll
  for (int r = 0; r < 4; ++r) {
    f[2*r]     = p[2*r];
    f[2*r+1]   = p[2*r+1];
    f[8+2*r]   = p[16 + 2*r];
    f[8+2*r+1] = p[16 + 2*r+1];
  }
  return f;
}

__device__ __forceinline__ v8f wmma_bf16(v16bf a, v16bf b, v8f c) {
  return __builtin_amdgcn_wmma_f32_16x16x32_bf16(false, a, false, b, (short)0, c,
                                                 false, false);
}

// ---------------- tiny prep kernels ----------------
__global__ void cast_f32_bf16(const float* __restrict__ src, bf16_t* __restrict__ dst, int n) {
  int i = blockIdx.x * 256 + threadIdx.x;
  if (i < n) dst[i] = f2bf(src[i]);
}

__global__ void bn_fold(const float* __restrict__ g, const float* __restrict__ b,
                        const float* __restrict__ m, const float* __restrict__ v,
                        float* __restrict__ sa, float* __restrict__ sc, int n) {
  int i = blockIdx.x * 128 + threadIdx.x;
  if (i < n) {
    float inv = g[i] * rsqrtf(v[i] + EPS_F);
    sa[i] = inv;
    sc[i] = b[i] - m[i] * inv;
  }
}

// ---------------- q = x^T @ q_w^T ----------------
// x: [B,C,N] f32, qw: [C,C] bf16 -> qout: [B*N, C] f32
__global__ void q_gemm(const float* __restrict__ x, const bf16_t* __restrict__ qw,
                       float* __restrict__ qout) {
  __shared__ bf16_t a_lds[16][Cc];
  int bx = blockIdx.x;
  int mtile = bx >> 1, halfc = bx & 1;
  int tid = threadIdx.x;
  int b = (mtile * 16) / Nn;
  int nbase = (mtile * 16) % Nn;
  int i = tid & 15, cb = tid >> 4;
#pragma unroll
  for (int pass = 0; pass < 16; ++pass) {
    int c = cb + pass * 16;
    a_lds[i][c] = f2bf(x[((size_t)b * Cc + c) * Nn + nbase + i]);
  }
  __syncthreads();
  int wave = tid >> 5, lane = tid & 31;
  int obase = halfc * 128 + wave * 16;
  v8f acc = zero8();
#pragma unroll
  for (int ks = 0; ks < 8; ++ks) {
    v16bf fa = frag_a(&a_lds[0][0], Cc, ks * 32);
    v16bf fb = frag_b(qw, Cc, obase, 1, ks * 32);
    acc = wmma_bf16(fa, fb, acc);
  }
  int col = obase + (lane & 15);
#pragma unroll
  for (int r = 0; r < 8; ++r) {
    int m = (lane < 16) ? r : r + 8;
    qout[(size_t)(mtile * 16 + m) * Cc + col] = acc[r];
  }
}

// ---------------- fused conv+BN+ReLU -> kv -> attention ----------------
// y: [B, CIN, N, K] f32. Per block: two positions g = b*N + n.
// aout: bf16 [B*N, 256], this scale writes columns [outoff, outoff+128).
template <int CIN>
__global__ void attn_fused(const float* __restrict__ y,
                           const bf16_t* __restrict__ convw,
                           const float* __restrict__ bna, const float* __restrict__ bnc,
                           const bf16_t* __restrict__ kvw,
                           const float* __restrict__ qf, int qoff,
                           bf16_t* __restrict__ aout, int outoff) {
  __shared__ float  yf[2][CIN][16];      // async-staged fp32, memory order [c][k]
  __shared__ bf16_t yt[2][16][CSs];      // repacked bf16 A tile [k][c]
  __shared__ bf16_t zt[2][16][CSs];
  __shared__ float  qs[2][CSs];
  int tid = threadIdx.x;
  int g0 = blockIdx.x * 2;

  // ---- async-stage both 16 x CIN y slices into LDS (16B per lane per transfer) ----
  {
    const int TOTAL = 2 * CIN * 4;       // number of 16-byte transfers
#pragma unroll
    for (int it = 0; it < TOTAL / 256; ++it) {
      int i = tid + 256 * it;
      int j = i / (CIN * 4);
      int r = i - j * (CIN * 4);
      int c = r >> 2, kq = r & 3;
      int g = g0 + j;
      unsigned b = (unsigned)(g >> 13);
      unsigned n = (unsigned)(g & (Nn - 1));
      unsigned goff = (((b * CIN + (unsigned)c) * Nn + n) * Kk + (unsigned)kq * 4u) * 4u;
      unsigned loff = (unsigned)(uintptr_t)&yf[j][c][kq * 4];
      async_copy16(loff, y, goff);
    }
  }
  // q rows for both positions
  {
    int j = tid >> 7, i = tid & 127;
    qs[j][i] = qf[(size_t)(g0 + j) * Cc + qoff + i];
  }
  if (tid < 32 && g0 + 2 < Bx * Nn) {    // speculative prefetch of next block's y slice
    int g = g0 + 2;
    int b = g >> 13, n = g & (Nn - 1);
    __builtin_prefetch(y + ((size_t)(b * CIN + (tid & (CIN - 1))) * Nn + n) * Kk, 0, 1);
  }
  wait_async0();
  __syncthreads();

  // ---- repack fp32 [c][k] -> bf16 [k][c] (WMMA A layout) ----
  {
    int kk = tid & 15, cb = tid >> 4;
#pragma unroll
    for (int j2 = 0; j2 < 2; ++j2)
#pragma unroll
      for (int p = 0; p < (CIN >> 4); ++p) {
        int c = cb + p * 16;
        yt[j2][kk][c] = f2bf(yf[j2][c][kk]);
      }
  }
  __syncthreads();

  int wave = tid >> 5, lane = tid & 31;
  int j = wave >> 2;

  // ---- conv 1x1 + BN + ReLU: z[16 x 128] ----
  {
    int ct = wave & 3;
    int ob = ct * 32;
    v8f a0 = zero8(), a1 = zero8();
#pragma unroll
    for (int ks = 0; ks < (CIN >> 5); ++ks) {
      v16bf fa = frag_a(&yt[j][0][0], CSs, ks * 32);
      a0 = wmma_bf16(fa, frag_b(convw, CIN, ob,      1, ks * 32), a0);
      a1 = wmma_bf16(fa, frag_b(convw, CIN, ob + 16, 1, ks * 32), a1);
    }
    int c0 = ob + (lane & 15), c1 = c0 + 16;
    float s0 = bna[c0], t0 = bnc[c0];
    float s1 = bna[c1], t1 = bnc[c1];
#pragma unroll
    for (int r = 0; r < 8; ++r) {
      int m = (lane < 16) ? r : r + 8;
      zt[j][m][c0] = f2bf(fmaxf(a0[r] * s0 + t0, 0.0f));
      zt[j][m][c1] = f2bf(fmaxf(a1[r] * s1 + t1, 0.0f));
    }
  }
  __syncthreads();

  // ---- kv projection (per-head, interleaved k/v rows) + attention ----
  {
    int h = wave & 3;
    int rb = h * 64;   // kv_w rows: h*64 + 2d (k), h*64 + 2d + 1 (v)
    v8f ak0 = zero8(), ak1 = zero8(), av0 = zero8(), av1 = zero8();
#pragma unroll
    for (int ks = 0; ks < 4; ++ks) {
      v16bf fa = frag_a(&zt[j][0][0], CSs, ks * 32);
      ak0 = wmma_bf16(fa, frag_b(kvw, CSs, rb,      2, ks * 32), ak0);
      ak1 = wmma_bf16(fa, frag_b(kvw, CSs, rb + 32, 2, ks * 32), ak1);
      av0 = wmma_bf16(fa, frag_b(kvw, CSs, rb + 1,  2, ks * 32), av0);
      av1 = wmma_bf16(fa, frag_b(kvw, CSs, rb + 33, 2, ks * 32), av1);
    }
    // acc layout: reg r, lane half -> neighbor m = r (+8), d = lane%16 (tile0) / 16+lane%16 (tile1)
    int l15 = lane & 15;
    float qlo = qs[j][h * 32 + l15];
    float qhi = qs[j][h * 32 + 16 + l15];
    float lr[8];
#pragma unroll
    for (int r = 0; r < 8; ++r) {
      float p = qlo * ak0[r] + qhi * ak1[r];
      p += __shfl_xor(p, 1, 32);
      p += __shfl_xor(p, 2, 32);
      p += __shfl_xor(p, 4, 32);
      p += __shfl_xor(p, 8, 32);          // logit for k = r + 8*(lane>=16)
      lr[r] = p * SCALE_F;
    }
    float lo[8];
#pragma unroll
    for (int r = 0; r < 8; ++r) lo[r] = __shfl_xor(lr[r], 16, 32);  // other half's k's
    float mx = -3.4e38f;
#pragma unroll
    for (int r = 0; r < 8; ++r) mx = fmaxf(mx, fmaxf(lr[r], lo[r]));
    float den = 0.0f, e[8];
#pragma unroll
    for (int r = 0; r < 8; ++r) {
      e[r] = __expf(lr[r] - mx);
      den += e[r] + __expf(lo[r] - mx);
    }
    float inv = 1.0f / den;
    float olo = 0.0f, ohi = 0.0f;
#pragma unroll
    for (int r = 0; r < 8; ++r) {
      float a = e[r] * inv;                // attn weight for this half's k = r+8*half
      olo += a * av0[r];
      ohi += a * av1[r];
    }
    olo += __shfl_xor(olo, 16, 32);        // combine both k-halves
    ohi += __shfl_xor(ohi, 16, 32);
    if (lane < 16) {
      size_t g = (size_t)g0 + j;
      aout[g * Cc + outoff + h * 32 + l15]       = f2bf(olo);
      aout[g * Cc + outoff + h * 32 + 16 + l15]  = f2bf(ohi);
    }
  }
}

// ---------------- out = concat @ proj_w^T + b, transposed to [B,C,N] ----------------
__global__ void proj_gemm(const bf16_t* __restrict__ ain, const bf16_t* __restrict__ pw,
                          const float* __restrict__ pb, float* __restrict__ out) {
  __shared__ float ot[16][128];           // [n within tile][co within half]
  int bx = blockIdx.x;
  int mtile = bx >> 1, halfc = bx & 1;
  int tid = threadIdx.x, wave = tid >> 5, lane = tid & 31;
  int obase = halfc * 128 + wave * 16;
  const bf16_t* abase = ain + (size_t)mtile * 16 * Cc;   // rows contiguous: gather direct
  v8f acc = zero8();
#pragma unroll
  for (int ks = 0; ks < 8; ++ks) {
    v16bf fa = frag_a(abase, Cc, ks * 32);
    v16bf fb = frag_b(pw, Cc, obase, 1, ks * 32);
    acc = wmma_bf16(fa, fb, acc);
  }
  int l15 = lane & 15;
  float bias = pb[obase + l15];
  int lcol = wave * 16 + l15;
#pragma unroll
  for (int r = 0; r < 8; ++r) {
    int m = (lane < 16) ? r : r + 8;
    ot[m][lcol] = acc[r] + bias;
  }
  __syncthreads();
  // transposed store: 128 co rows x 16 consecutive n, two float4 per thread
  int g0 = mtile * 16;
  int b = g0 >> 13, n0 = g0 & (Nn - 1);
  int co = tid & 127;
  int q4 = tid >> 7;                      // 0..1
  float* orow = out + ((size_t)b * Cc + halfc * 128 + co) * Nn + n0;
#pragma unroll
  for (int t = 0; t < 2; ++t) {
    int nq = q4 * 2 + t;                  // 0..3
    float4 vv = make_float4(ot[nq*4+0][co], ot[nq*4+1][co],
                            ot[nq*4+2][co], ot[nq*4+3][co]);
    *(float4*)(orow + nq * 4) = vv;
  }
}

// ---------------- host launch ----------------
extern "C" void kernel_launch(void* const* d_in, const int* in_sizes, int n_in,
                              void* d_out, int out_size, void* d_ws, size_t ws_size,
                              hipStream_t stream) {
  const float* x       = (const float*)d_in[0];
  const float* y0      = (const float*)d_in[1];
  const float* y1      = (const float*)d_in[2];
  const float* q_w     = (const float*)d_in[3];
  const float* conv0_w = (const float*)d_in[4];
  const float* bn0_g   = (const float*)d_in[5];
  const float* bn0_b   = (const float*)d_in[6];
  const float* bn0_m   = (const float*)d_in[7];
  const float* bn0_v   = (const float*)d_in[8];
  const float* kv0_w   = (const float*)d_in[9];
  const float* conv1_w = (const float*)d_in[10];
  const float* bn1_g   = (const float*)d_in[11];
  const float* bn1_b   = (const float*)d_in[12];
  const float* bn1_m   = (const float*)d_in[13];
  const float* bn1_v   = (const float*)d_in[14];
  const float* kv1_w   = (const float*)d_in[15];
  const float* proj_w  = (const float*)d_in[16];
  const float* proj_b  = (const float*)d_in[17];
  float* out = (float*)d_out;

  char* ws = (char*)d_ws;
  size_t off = 0;
  auto alloc = [&](size_t bytes) { char* p = ws + off; off = (off + bytes + 255) & ~(size_t)255; return p; };
  bf16_t* qw_bf    = (bf16_t*)alloc(Cc * Cc * 2);
  bf16_t* conv0_bf = (bf16_t*)alloc(CSs * 64 * 2);
  bf16_t* conv1_bf = (bf16_t*)alloc(CSs * 128 * 2);
  bf16_t* kv0_bf   = (bf16_t*)alloc(256 * CSs * 2);
  bf16_t* kv1_bf   = (bf16_t*)alloc(256 * CSs * 2);
  bf16_t* proj_bf  = (bf16_t*)alloc(Cc * Cc * 2);
  float*  bn0_a    = (float*)alloc(CSs * 4);
  float*  bn0_c    = (float*)alloc(CSs * 4);
  float*  bn1_a    = (float*)alloc(CSs * 4);
  float*  bn1_c    = (float*)alloc(CSs * 4);
  float*  q_f32    = (float*)alloc((size_t)Bx * Nn * Cc * 4);      // 33.5 MB
  bf16_t* attn_bf  = (bf16_t*)alloc((size_t)Bx * Nn * Cc * 2);     // 16.8 MB
  (void)ws_size; (void)in_sizes; (void)n_in; (void)out_size;

  // weight prep
  bn_fold<<<1, 128, 0, stream>>>(bn0_g, bn0_b, bn0_m, bn0_v, bn0_a, bn0_c, CSs);
  bn_fold<<<1, 128, 0, stream>>>(bn1_g, bn1_b, bn1_m, bn1_v, bn1_a, bn1_c, CSs);
  cast_f32_bf16<<<(Cc * Cc + 255) / 256, 256, 0, stream>>>(q_w, qw_bf, Cc * Cc);
  cast_f32_bf16<<<(CSs * 64 + 255) / 256, 256, 0, stream>>>(conv0_w, conv0_bf, CSs * 64);
  cast_f32_bf16<<<(CSs * 128 + 255) / 256, 256, 0, stream>>>(conv1_w, conv1_bf, CSs * 128);
  cast_f32_bf16<<<(256 * CSs + 255) / 256, 256, 0, stream>>>(kv0_w, kv0_bf, 256 * CSs);
  cast_f32_bf16<<<(256 * CSs + 255) / 256, 256, 0, stream>>>(kv1_w, kv1_bf, 256 * CSs);
  cast_f32_bf16<<<(Cc * Cc + 255) / 256, 256, 0, stream>>>(proj_w, proj_bf, Cc * Cc);

  // q projection
  q_gemm<<<(Bx * Nn / 16) * 2, 256, 0, stream>>>(x, qw_bf, q_f32);

  // fused attention per scale
  attn_fused<64><<<Bx * Nn / 2, 256, 0, stream>>>(y0, conv0_bf, bn0_a, bn0_c, kv0_bf,
                                                  q_f32, 0, attn_bf, 0);
  attn_fused<128><<<Bx * Nn / 2, 256, 0, stream>>>(y1, conv1_bf, bn1_a, bn1_c, kv1_bf,
                                                   q_f32, 128, attn_bf, 128);

  // output projection (+ transpose to [B,C,N])
  proj_gemm<<<(Bx * Nn / 16) * 2, 256, 0, stream>>>(attn_bf, proj_bf, proj_b, out);
}
